// GATModel_8117488190228
// MI455X (gfx1250) — matrix operand order repaired and checked by reading, not compiled, not run on hardware
//
#include <hip/hip_runtime.h>
#include <hip/hip_bf16.h>
#include <math.h>

// ---------------------------------------------------------------------------
// Problem constants (from the reference)
// ---------------------------------------------------------------------------
#define Nn     1024      // nodes per graph
#define Ff     128       // input features
#define Hh     8         // heads
#define Cc     32        // channels per head
#define Dd     256       // H*C
#define Ee     32768     // edges
#define NTOT   (Ee + Nn) // edges + self loops
#define D4     1024      // 4*D (LSTM gate width)
#define NEG_SLOPE 0.2f

typedef __attribute__((ext_vector_type(16))) _Float16 v16h;
typedef __attribute__((ext_vector_type(8)))  float    v8f;

struct alignas(16) H8 { _Float16 h[8]; };

// ---------------------------------------------------------------------------
// Helpers: order-preserving float<->uint for atomic max
// ---------------------------------------------------------------------------
__device__ __forceinline__ unsigned ordf(float f) {
    unsigned u = __float_as_uint(f);
    return (u & 0x80000000u) ? ~u : (u | 0x80000000u);
}
__device__ __forceinline__ float deordf(unsigned o) {
    unsigned u = (o & 0x80000000u) ? (o & 0x7fffffffu) : ~o;
    return __uint_as_float(u);
}
__device__ __forceinline__ float sigm(float x) { return 1.0f / (1.0f + __expf(-x)); }

// ---------------------------------------------------------------------------
// f32 -> f16 elementwise convert
// ---------------------------------------------------------------------------
__global__ void conv16(const float* __restrict__ src, _Float16* __restrict__ dst, int n) {
    int i = blockIdx.x * blockDim.x + threadIdx.x;
    if (i < n) dst[i] = (_Float16)src[i];
}

// f32 (K x N) -> f16 transposed (N x K)
__global__ void convT16(const float* __restrict__ src, _Float16* __restrict__ dst, int K, int N) {
    int i = blockIdx.x * blockDim.x + threadIdx.x;
    if (i >= K * N) return;
    int n = i / K, k = i - n * K;
    dst[i] = (_Float16)src[(size_t)k * N + n];
}

// f32 (R x C) -> f32 transposed (C x R)   (for Whh -> WhhT)
__global__ void convT32(const float* __restrict__ src, float* __restrict__ dst, int R, int C) {
    int i = blockIdx.x * blockDim.x + threadIdx.x;
    if (i >= R * C) return;
    int c = i / R, r = i - c * R;          // dst[c*R + r] = src[r*C + c]
    dst[i] = src[(size_t)r * C + c];
}

// ---------------------------------------------------------------------------
// Per-lane fragment loaders per CDNA5 ISA 7.12.2 (wave32).
// A (16x32 f16): lane<16 holds K {kb..kb+7} U {kb+16..kb+23}; lane>=16 +8.
// B (32x16 f16): same K order as A (lane<16 K {kb..kb+7} U {kb+16..kb+23}; lane>=16 +8),
//                read from B stored transposed (N x K).
// ---------------------------------------------------------------------------
__device__ __forceinline__ v16h load_fragA(const _Float16* Arow, int kb, int laneHi) {
    H8 lo = *(const H8*)(Arow + kb + 8 * laneHi);
    H8 hi = *(const H8*)(Arow + kb + 8 * laneHi + 16);
    v16h v;
#pragma unroll
    for (int i = 0; i < 8; ++i) { v[i] = lo.h[i]; v[i + 8] = hi.h[i]; }
    return v;
}
__device__ __forceinline__ v16h load_fragB(const _Float16* Brow, int kb, int laneHi) {
    H8 lo = *(const H8*)(Brow + kb + 8 * laneHi);
    H8 hi = *(const H8*)(Brow + kb + 8 * laneHi + 16);
    v16h v;
#pragma unroll
    for (int i = 0; i < 8; ++i) { v[i] = lo.h[i]; v[i + 8] = hi.h[i]; }
    return v;
}

// ---------------------------------------------------------------------------
// WMMA GEMM:  C(MxN,f32) = A(MxK,f16 row-major) * B, with B given transposed
// as Bt(NxK,f16 row-major).  2x2 register tiling: each wave owns a 32x32
// output block (4 accumulators) so each A/B fragment feeds 2 WMMAs.
// 4 waves per block -> block covers 32(M) x 128(N).
// ---------------------------------------------------------------------------
__global__ void wmma_gemm(const _Float16* __restrict__ A,
                          const _Float16* __restrict__ Bt,
                          float* __restrict__ C,
                          int M, int N, int K)
{
    const int lane   = threadIdx.x & 31;
    const int wave   = threadIdx.x >> 5;
    const int tileN  = blockIdx.x * 4 + wave;   // 32-wide N tile index
    const int tileM  = blockIdx.y;              // 32-wide M tile index
    if (tileN * 32 >= N || tileM * 32 >= M) return;

    const int laneHi = lane >> 4;      // 0 or 1
    const int lane15 = lane & 15;
    const int m0 = tileM * 32 + lane15;
    const int n0 = tileN * 32 + lane15;

    const _Float16* Arow0 = A  + (size_t)m0 * K;
    const _Float16* Arow1 = Arow0 + (size_t)16 * K;
    const _Float16* Brow0 = Bt + (size_t)n0 * K;
    const _Float16* Brow1 = Brow0 + (size_t)16 * K;

    v8f acc00 = {}, acc01 = {}, acc10 = {}, acc11 = {};
    for (int kb = 0; kb < K; kb += 32) {
        v16h a0 = load_fragA(Arow0, kb, laneHi);
        v16h a1 = load_fragA(Arow1, kb, laneHi);
        v16h b0 = load_fragB(Brow0, kb, laneHi);
        v16h b1 = load_fragB(Brow1, kb, laneHi);
        acc00 = __builtin_amdgcn_wmma_f32_16x16x32_f16(false, a0, false, b0, (short)0, acc00, false, false);
        acc01 = __builtin_amdgcn_wmma_f32_16x16x32_f16(false, a0, false, b1, (short)0, acc01, false, false);
        acc10 = __builtin_amdgcn_wmma_f32_16x16x32_f16(false, a1, false, b0, (short)0, acc10, false, false);
        acc11 = __builtin_amdgcn_wmma_f32_16x16x32_f16(false, a1, false, b1, (short)0, acc11, false, false);
    }
    // C/D layout: VGPR v -> row (v + 8*laneHi), col (lane&15)
#pragma unroll
    for (int v = 0; v < 8; ++v) {
        int r0 = tileM * 32 + v + 8 * laneHi;
        int r1 = r0 + 16;
        C[(size_t)r0 * N + n0]      = acc00[v];
        C[(size_t)r0 * N + n0 + 16] = acc01[v];
        C[(size_t)r1 * N + n0]      = acc10[v];
        C[(size_t)r1 * N + n0 + 16] = acc11[v];
    }
}

// ---------------------------------------------------------------------------
// GAT attention pieces
// ---------------------------------------------------------------------------
// es[n,h] = sum_c h[n,h,c]*a_src[h,c] ; ed likewise
__global__ void att_scores(const float* __restrict__ hlin,
                           const float* __restrict__ asrc,
                           const float* __restrict__ adst,
                           float* __restrict__ es, float* __restrict__ ed)
{
    int idx = blockIdx.x * blockDim.x + threadIdx.x;     // N*H
    if (idx >= Nn * Hh) return;
    int node = idx >> 3, head = idx & 7;
    const float* hp = hlin + (size_t)node * Dd + head * Cc;
    const float* ap = asrc + head * Cc;
    const float* dp = adst + head * Cc;
    float s = 0.f, d = 0.f;
#pragma unroll 8
    for (int c = 0; c < Cc; ++c) { s += hp[c] * ap[c]; d += hp[c] * dp[c]; }
    es[idx] = s; ed[idx] = d;
}

// per (edge,head): leaky_relu score; segment max over dst via atomic ordered max
__global__ void edge_logits(const int* __restrict__ ei,
                            const float* __restrict__ es, const float* __restrict__ ed,
                            float* __restrict__ score, unsigned* __restrict__ mord)
{
    int idx = blockIdx.x * blockDim.x + threadIdx.x;     // NTOT*H
    if (idx >= NTOT * Hh) return;
    int e = idx >> 3, head = idx & 7;
    int s, d;
    if (e < Ee) { s = ei[e]; d = ei[Ee + e]; } else { s = d = e - Ee; }
    float v = es[s * Hh + head] + ed[d * Hh + head];
    v = (v > 0.f) ? v : NEG_SLOPE * v;
    score[idx] = v;
    atomicMax(mord + d * Hh + head, ordf(v));
}

// p = exp(score - m[dst]); segment-sum denominator
__global__ void edge_exp(const int* __restrict__ ei,
                         float* __restrict__ score,
                         const unsigned* __restrict__ mord,
                         float* __restrict__ denom)
{
    int idx = blockIdx.x * blockDim.x + threadIdx.x;     // NTOT*H
    if (idx >= NTOT * Hh) return;
    int e = idx >> 3, head = idx & 7;
    int d = (e < Ee) ? ei[Ee + e] : (e - Ee);
    float m = deordf(mord[d * Hh + head]);
    if (!(m > -3.0e38f && m < 3.0e38f)) m = 0.f;         // where(isfinite(m), m, 0)
    float p = __expf(score[idx] - m);
    score[idx] = p;                                      // in place: score -> pnum
    atomicAdd(denom + d * Hh + head, p);
}

// out[dst,h,c] += (p/denom[dst,h]) * hlin[src,h,c]
__global__ void aggregate(const int* __restrict__ ei,
                          const float* __restrict__ pnum,
                          const float* __restrict__ denom,
                          const float* __restrict__ hlin,
                          float* __restrict__ outp)
{
    int idx = blockIdx.x * blockDim.x + threadIdx.x;     // NTOT*H*C
    if (idx >= NTOT * Dd) return;
    int c = idx & 31, head = (idx >> 5) & 7, e = idx >> 8;
    int s, d;
    if (e < Ee) { s = ei[e]; d = ei[Ee + e]; } else { s = d = e - Ee; }
    float alpha = pnum[e * Hh + head] / (denom[d * Hh + head] + 1e-16f);
    atomicAdd(outp + (size_t)d * Dd + head * Cc + c,
              alpha * hlin[(size_t)s * Dd + head * Cc + c]);
}

// relu(agg + b) -> f16 (feeds the next WMMA GEMM as A)
__global__ void bias_relu_h16(const float* __restrict__ agg, const float* __restrict__ b,
                              _Float16* __restrict__ out16)
{
    int idx = blockIdx.x * blockDim.x + threadIdx.x;     // N*D
    if (idx >= Nn * Dd) return;
    float v = agg[idx] + b[idx & (Dd - 1)];
    out16[idx] = (_Float16)(v > 0.f ? v : 0.f);
}

// ---------------------------------------------------------------------------
// Persistent single-block LSTM over 1024 sequential steps, batch 1.
// ZX[t, :] = hh[t] @ Wih.T precomputed by WMMA GEMM. WhhT is (256 x 1024).
// ---------------------------------------------------------------------------
__global__ void __launch_bounds__(1024)
lstm_seq(const float* __restrict__ ZX, const float* __restrict__ WhhT,
         const float* __restrict__ bih, const float* __restrict__ bhh,
         float* __restrict__ hs)
{
    __shared__ float hbuf[Dd];
    __shared__ float zbuf[D4];
    const int tid = threadIdx.x;                  // 1024 threads
    float cstate = 0.f;
    if (tid < Dd) hbuf[tid] = 0.f;
    const float bias = bih[tid] + bhh[tid];
    __syncthreads();

    for (int t = 0; t < Nn; ++t) {
        // hide next step's ZX fetch behind this step's 256-MAC dot
        if (t + 1 < Nn) __builtin_prefetch(&ZX[(size_t)(t + 1) * D4 + tid], 0, 0);
        float acc = ZX[(size_t)t * D4 + tid] + bias;
#pragma unroll 8
        for (int k = 0; k < Dd; ++k)
            acc += hbuf[k] * WhhT[(size_t)k * D4 + tid];   // coalesced across lanes
        zbuf[tid] = acc;
        __syncthreads();
        if (tid < Dd) {
            float ig = sigm(zbuf[tid]);
            float fg = sigm(zbuf[tid + Dd]);
            float gg = tanhf(zbuf[tid + 2 * Dd]);
            float og = sigm(zbuf[tid + 3 * Dd]);
            cstate = fg * cstate + ig * gg;
            float hc = og * tanhf(cstate);
            hbuf[tid] = hc;
            hs[(size_t)t * Dd + tid] = hc;
        }
        __syncthreads();
    }
}

// ---------------------------------------------------------------------------
// Final heads: xe = hs[src]*hs[dst]; pred = xe.Wp + bp ; logit = xe.Wc + bc
// One wave per edge, wave32 shuffle reduction.
// ---------------------------------------------------------------------------
__global__ void edge_head(const float* __restrict__ hs, const int* __restrict__ ei,
                          const float* __restrict__ Wp, const float* __restrict__ bp,
                          const float* __restrict__ Wc, const float* __restrict__ bc,
                          float* __restrict__ out)
{
    const int lane = threadIdx.x & 31;
    const int wave = threadIdx.x >> 5;
    const int e = blockIdx.x * (blockDim.x >> 5) + wave;
    if (e >= Ee) return;
    const float* hsrc = hs + (size_t)ei[e] * Dd;
    const float* hdst = hs + (size_t)ei[Ee + e] * Dd;
    float ap = 0.f, ac = 0.f;
    for (int c = lane; c < Dd; c += 32) {
        float xe = hsrc[c] * hdst[c];
        ap += xe * Wp[c];
        ac += xe * Wc[c];
    }
    for (int off = 16; off; off >>= 1) {
        ap += __shfl_down(ap, off, 32);
        ac += __shfl_down(ac, off, 32);
    }
    if (lane == 0) {
        out[e]      = ap + bp[0];    // edge_weights_pred (E,1)
        out[Ee + e] = ac + bc[0];    // logits (E,)
    }
}

// ---------------------------------------------------------------------------
// Host: launch sequence (all on `stream`, scratch from d_ws)
// ---------------------------------------------------------------------------
extern "C" void kernel_launch(void* const* d_in, const int* in_sizes, int n_in,
                              void* d_out, int out_size, void* d_ws, size_t ws_size,
                              hipStream_t stream)
{
    const float* x    = (const float*)d_in[0];   // (B,T,N,F) -> need only x[0,0]
    const int*   ei   = (const int*)  d_in[1];   // (2,E)
    const float* W1   = (const float*)d_in[3];
    const float* as1  = (const float*)d_in[4];
    const float* ad1  = (const float*)d_in[5];
    const float* b1   = (const float*)d_in[6];
    const float* W2   = (const float*)d_in[7];
    const float* as2  = (const float*)d_in[8];
    const float* ad2  = (const float*)d_in[9];
    const float* b2   = (const float*)d_in[10];
    const float* Wih  = (const float*)d_in[11];  // (1024,256)
    const float* Whh  = (const float*)d_in[12];  // (1024,256)
    const float* bih  = (const float*)d_in[13];
    const float* bhh  = (const float*)d_in[14];
    const float* Wp   = (const float*)d_in[15];
    const float* bp   = (const float*)d_in[16];
    const float* Wc   = (const float*)d_in[17];
    const float* bc   = (const float*)d_in[18];
    float* out = (float*)d_out;

    char* ws = (char*)d_ws;
    size_t off = 0;
    auto take = [&](size_t bytes) { char* p = ws + off; off += (bytes + 255) & ~size_t(255); return p; };

    float*     hlin  = (float*)    take(sizeof(float) * Nn * Dd);       // GEMM output (per layer)
    float*     agg   = (float*)    take(sizeof(float) * Nn * Dd);       // aggregation accumulator
    float*     score = (float*)    take(sizeof(float) * NTOT * Hh);     // scores then pnum
    unsigned*  mord  = (unsigned*) take(sizeof(unsigned) * Nn * Hh);
    float*     denom = (float*)    take(sizeof(float) * Nn * Hh);
    float*     es    = (float*)    take(sizeof(float) * Nn * Hh);
    float*     ed    = (float*)    take(sizeof(float) * Nn * Hh);
    _Float16*  A16   = (_Float16*) take(sizeof(_Float16) * Nn * Dd);    // f16 A for GEMMs
    _Float16*  W1t16 = (_Float16*) take(sizeof(_Float16) * Dd * Ff);    // (256 x 128)
    _Float16*  W2t16 = (_Float16*) take(sizeof(_Float16) * Dd * Dd);    // (256 x 256)
    _Float16*  Wih16 = (_Float16*) take(sizeof(_Float16) * D4 * Dd);    // (1024 x 256) = (Wih.T) transposed
    float*     WhhT  = (float*)    take(sizeof(float) * Dd * D4);       // (256 x 1024)
    float*     ZX    = (float*)    take(sizeof(float) * Nn * D4);       // (1024 x 1024)
    float*     hs    = (float*)    take(sizeof(float) * Nn * Dd);       // LSTM outputs

    const int TB = 256;
    auto blocks = [](int n, int tb) { return (n + tb - 1) / tb; };

    // ---- one-time conversions -------------------------------------------
    conv16 <<<blocks(Nn * Ff, TB), TB, 0, stream>>>(x, A16, Nn * Ff);         // X[0,0] -> f16
    convT16<<<blocks(Ff * Dd, TB), TB, 0, stream>>>(W1, W1t16, Ff, Dd);
    convT16<<<blocks(Dd * Dd, TB), TB, 0, stream>>>(W2, W2t16, Dd, Dd);
    conv16 <<<blocks(D4 * Dd, TB), TB, 0, stream>>>(Wih, Wih16, D4 * Dd);     // Bt of Wih.T == Wih
    convT32<<<blocks(D4 * Dd, TB), TB, 0, stream>>>(Whh, WhhT, D4, Dd);

    // ---- GAT layers ------------------------------------------------------
    for (int layer = 0; layer < 2; ++layer) {
        const _Float16* Bt = layer ? W2t16 : W1t16;
        const int K        = layer ? Dd : Ff;
        const float *asrc = layer ? as2 : as1, *adst = layer ? ad2 : ad1, *bias = layer ? b2 : b1;

        dim3 g(Dd / 128, Nn / 32);                 // waves tile 32x32 outputs
        wmma_gemm<<<g, 128, 0, stream>>>(A16, Bt, hlin, Nn, Dd, K);

        att_scores<<<blocks(Nn * Hh, TB), TB, 0, stream>>>(hlin, asrc, adst, es, ed);

        hipMemsetAsync(mord,  0, sizeof(unsigned) * Nn * Hh, stream);
        hipMemsetAsync(denom, 0, sizeof(float)    * Nn * Hh, stream);
        hipMemsetAsync(agg,   0, sizeof(float)    * Nn * Dd, stream);

        edge_logits<<<blocks(NTOT * Hh, TB), TB, 0, stream>>>(ei, es, ed, score, mord);
        edge_exp   <<<blocks(NTOT * Hh, TB), TB, 0, stream>>>(ei, score, mord, denom);
        aggregate  <<<blocks(NTOT * Dd, TB), TB, 0, stream>>>(ei, score, denom, hlin, agg);
        bias_relu_h16<<<blocks(Nn * Dd, TB), TB, 0, stream>>>(agg, bias, A16);
    }

    // ---- LSTM pre-projection: ZX = hh @ Wih.T  (1024x256 * 256x1024) -----
    {
        dim3 g(D4 / 128, Nn / 32);
        wmma_gemm<<<g, 128, 0, stream>>>(A16, Wih16, ZX, Nn, D4, Dd);
    }

    // ---- serial LSTM (single persistent block, 1024 steps) ---------------
    lstm_seq<<<1, 1024, 0, stream>>>(ZX, WhhT, bih, bhh, hs);

    // ---- edge heads ------------------------------------------------------
    edge_head<<<blocks(Ee * 32, TB), TB, 0, stream>>>(hs, ei, Wp, bp, Wc, bc, out);

    (void)in_sizes; (void)n_in; (void)out_size; (void)ws_size;
}
